// GCNModel_11579231830751
// MI455X (gfx1250) — matrix operand order repaired and checked by reading, not compiled, not run on hardware
//
#include <hip/hip_runtime.h>
#include <hip/hip_bf16.h>
#include <math.h>

// ---------------------------------------------------------------------------
// Problem constants (from reference)
// ---------------------------------------------------------------------------
#define Nn 8192
#define Dd 256
#define Hh 256
#define Cc 64
#define Ee 262144
#define Mm 2048
#define TAUf 0.5f
#define HP1f 0.9f
#define WDf  5e-4f

typedef __attribute__((ext_vector_type(16))) _Float16 v16h;
typedef __attribute__((ext_vector_type(8)))  float    v8f;
typedef __attribute__((ext_vector_type(4)))  int      v4i;

#define GAS __attribute__((address_space(1)))
#define LAS __attribute__((address_space(3)))

#if defined(__AMDGCN__) && __has_builtin(__builtin_amdgcn_global_load_async_to_lds_b128)
#define HAVE_ASYNC 1
#endif

// 16-byte global->LDS copy: async engine if available, else through VGPRs
__device__ __forceinline__ void async_cp16(const _Float16* g, _Float16* l) {
#if defined(HAVE_ASYNC)
    __builtin_amdgcn_global_load_async_to_lds_b128(
        (GAS v4i*)(g), (LAS v4i*)(l), 0, 0);
#else
    *(uint4*)l = *(const uint4*)g;
#endif
}

__device__ __forceinline__ void async_wait0() {
#if defined(HAVE_ASYNC)
#if __has_builtin(__builtin_amdgcn_s_wait_asynccnt)
    __builtin_amdgcn_s_wait_asynccnt(0);
#else
    asm volatile("s_wait_asynccnt 0" ::: "memory");
#endif
#endif
}

// ---------------------------------------------------------------------------
// f32 -> f16 conversion
// ---------------------------------------------------------------------------
__global__ void k_cvt_f16(const float* __restrict__ s, _Float16* __restrict__ d, long n) {
    long i = (long)blockIdx.x * blockDim.x + threadIdx.x;
    if (i < n) d[i] = (_Float16)s[i];
}

// ---------------------------------------------------------------------------
// WMMA f16 GEMM: C[M,N] = A[M,K] @ B[K,N] + bias (+relu), f32 accumulate
// Block tile 128x64, 8 waves (4x2), each wave 32x32 = 2x2 WMMA 16x16x32 tiles
// A tile staged via async global->LDS b128; B tile staged transposed so both
// fragment gathers are pure ds_load_b128 pairs.
// ---------------------------------------------------------------------------
#define BM 128
#define BN 64
#define BK 32
#define BSTRIDE 40   // transposed-B LDS row stride in halfs (80B, 16B-aligned)

template <int RELU>
__global__ __launch_bounds__(256) void k_gemm_f16(
    const _Float16* __restrict__ A, const _Float16* __restrict__ B,
    const float* __restrict__ bias, float* __restrict__ C,
    int Md, int Nd, int Kd)
{
    __shared__ __align__(16) _Float16 As[BM * BK];
    __shared__ __align__(16) _Float16 Bst[BN * BSTRIDE];

    const int tid  = threadIdx.x;
    const int lane = tid & 31;
    const int wv   = tid >> 5;    // 0..7
    const int wm   = wv & 3;      // wave row within block (4)
    const int wn   = wv >> 2;     // wave col within block (2)
    const int bm   = blockIdx.y * BM;
    const int bn   = blockIdx.x * BN;

    v8f zero = {};
    v8f acc[2][2];
    acc[0][0] = zero; acc[0][1] = zero; acc[1][0] = zero; acc[1][1] = zero;

    // cooperative load indexing
    const int ar  = tid >> 1;           // 0..127 : A tile row
    const int ac  = (tid & 1) * 16;     // 0 or 16 : A tile col start
    const int br  = tid >> 3;           // 0..31  : B tile row (k)
    const int bc8 = (tid & 7) * 8;      // B tile col start (8 halfs along N)

    const int kbaseA = (lane >> 4) << 3;   // lanes 16..31 -> +8 in K
    const int kbaseB = (lane >> 4) << 3;   // lanes 16..31 -> +8 in K
    const int l15    = lane & 15;

    for (int kb = 0; kb < Kd; kb += BK) {
        // --- A tile 128x32: two async b128 copies per thread
        const _Float16* ag = A + (long)(bm + ar) * Kd + kb + ac;
        async_cp16(ag,     &As[ar * BK + ac]);
        async_cp16(ag + 8, &As[ar * BK + ac + 8]);

        // --- B tile 32x64, transposed into LDS: Bst[n][k]
        const _Float16* bg = B + (long)(kb + br) * Nd + bn + bc8;
        uint4 braw = *(const uint4*)(bg);
        const _Float16* bh = (const _Float16*)&braw;
        #pragma unroll
        for (int q = 0; q < 8; ++q)
            Bst[(bc8 + q) * BSTRIDE + br] = bh[q];

        __builtin_prefetch(ag + BK, 0, 1);   // global_prefetch next K panel
        async_wait0();                        // s_wait_asynccnt 0
        __syncthreads();

        // --- fragments: each is 2x ds_load_b128 (contiguous 8-half runs)
        v16h af[2], bf[2];
        #pragma unroll
        for (int mi = 0; mi < 2; ++mi) {
            const _Float16* ap = &As[(wm * 32 + mi * 16 + l15) * BK + kbaseA];
            ((uint4*)&af[mi])[0] = *(const uint4*)(ap);        // k base..base+7
            ((uint4*)&af[mi])[1] = *(const uint4*)(ap + 16);   // k base+16..+23
        }
        #pragma unroll
        for (int ni = 0; ni < 2; ++ni) {
            const _Float16* bp = &Bst[(wn * 32 + ni * 16 + l15) * BSTRIDE + kbaseB];
            ((uint4*)&bf[ni])[0] = *(const uint4*)(bp);        // k base..base+7
            ((uint4*)&bf[ni])[1] = *(const uint4*)(bp + 16);   // k base+16..+23
        }
        #pragma unroll
        for (int mi = 0; mi < 2; ++mi)
            #pragma unroll
            for (int ni = 0; ni < 2; ++ni)
                acc[mi][ni] = __builtin_amdgcn_wmma_f32_16x16x32_f16(
                    false, af[mi], false, bf[ni], (short)0, acc[mi][ni],
                    false, false);
        __syncthreads();
    }

    // Epilogue: C/D layout — lane<16: m=v, n=lane; lane>=16: m=v+8, n=lane-16
    const int mofs = (lane >> 4) << 3;
    const int ncol = lane & 15;
    #pragma unroll
    for (int ni = 0; ni < 2; ++ni) {
        const int col = bn + wn * 32 + ni * 16 + ncol;
        const float bv = bias[col];
        #pragma unroll
        for (int mi = 0; mi < 2; ++mi) {
            const int row0 = bm + wm * 32 + mi * 16 + mofs;
            float* cp = C + (long)row0 * Nd + col;
            #pragma unroll
            for (int v = 0; v < 8; ++v) {
                float x = acc[mi][ni][v] + bv;
                if (RELU) x = fmaxf(x, 0.0f);
                cp[(long)v * Nd] = x;
            }
        }
    }
}

// ---------------------------------------------------------------------------
// Row-wise L2 normalize (cols == blockDim.x == 64)
// ---------------------------------------------------------------------------
__global__ void k_l2norm(const float* __restrict__ in, float* __restrict__ out, int cols) {
    __shared__ float red[64];
    int r = blockIdx.x, t = threadIdx.x;
    float v = in[(long)r * cols + t];
    red[t] = v * v; __syncthreads();
    for (int s = 32; s > 0; s >>= 1) { if (t < s) red[t] += red[t + s]; __syncthreads(); }
    float n = fmaxf(sqrtf(red[0]), 1e-12f);
    out[(long)r * cols + t] = v / n;
}

// outputs = l2norm(0.6*g + 0.4*h)
__global__ void k_combine(const float* __restrict__ g, const float* __restrict__ h,
                          float* __restrict__ out, int cols) {
    __shared__ float red[64];
    int r = blockIdx.x, t = threadIdx.x;
    float v = 0.6f * g[(long)r * cols + t] + 0.4f * h[(long)r * cols + t];
    red[t] = v * v; __syncthreads();
    for (int s = 32; s > 0; s >>= 1) { if (t < s) red[t] += red[t + s]; __syncthreads(); }
    float n = fmaxf(sqrtf(red[0]), 1e-12f);
    out[(long)r * cols + t] = v / n;
}

__global__ void k_zero(float* p, int n) {
    int i = blockIdx.x * blockDim.x + threadIdx.x;
    if (i < n) p[i] = 0.0f;
}

__global__ void k_masksum(const int* __restrict__ mask, float* sc, int n) {
    __shared__ float red[256];
    int t = threadIdx.x; float s = 0.f;
    for (int i = blockIdx.x * 256 + t; i < n; i += gridDim.x * 256) s += (float)mask[i];
    red[t] = s; __syncthreads();
    for (int st = 128; st > 0; st >>= 1) { if (t < st) red[t] += red[t + st]; __syncthreads(); }
    if (t == 0) atomicAdd(&sc[0], red[0]);
}

// masked softmax CE + accuracy, block per row (64 threads)
__global__ void k_ce(const float* __restrict__ outp, const float* __restrict__ labels,
                     const int* __restrict__ mask, const float* __restrict__ sc,
                     float* lossAcc, float* accAcc, int cols, int rows) {
    __shared__ float red[64];
    __shared__ int  ired[64];
    int r = blockIdx.x, t = threadIdx.x;
    float o = outp[(long)r * cols + t];
    float l = labels[(long)r * cols + t];

    red[t] = o; __syncthreads();
    for (int s = 32; s > 0; s >>= 1) { if (t < s) red[t] = fmaxf(red[t], red[t + s]); __syncthreads(); }
    float mx = red[0]; __syncthreads();
    red[t] = expf(o - mx); __syncthreads();
    for (int s = 32; s > 0; s >>= 1) { if (t < s) red[t] += red[t + s]; __syncthreads(); }
    float lse = mx + logf(red[0]); __syncthreads();
    red[t] = l * (lse - o); __syncthreads();
    for (int s = 32; s > 0; s >>= 1) { if (t < s) red[t] += red[t + s]; __syncthreads(); }
    float ce = red[0]; __syncthreads();

    red[t] = o; ired[t] = t; __syncthreads();
    for (int s = 32; s > 0; s >>= 1) {
        if (t < s && red[t + s] > red[t]) { red[t] = red[t + s]; ired[t] = ired[t + s]; }
        __syncthreads();
    }
    int am_o = ired[0]; __syncthreads();
    red[t] = l; ired[t] = t; __syncthreads();
    for (int s = 32; s > 0; s >>= 1) {
        if (t < s && red[t + s] > red[t]) { red[t] = red[t + s]; ired[t] = ired[t + s]; }
        __syncthreads();
    }
    int am_l = ired[0];
    if (t == 0) {
        float maskf = (float)mask[r] * (float)rows / sc[0];
        atomicAdd(lossAcc, ce * maskf);
        atomicAdd(accAcc, (am_o == am_l ? 1.0f : 0.0f) * maskf);
    }
}

__device__ inline float logsig(float z) {
    return (z >= 0.f) ? -log1pf(expf(-z)) : (z - log1pf(expf(z)));
}

// edge pair loss: sum_e -(logsig(z1)+logsig(z2))
__global__ __launch_bounds__(256) void k_edge(
    const float* __restrict__ g, const float* __restrict__ h,
    const int* __restrict__ ei, const int* __restrict__ ej,
    const float* __restrict__ Wc, const float* __restrict__ bc,
    float* accum, int E, int cols)
{
    __shared__ float sW[128];
    __shared__ float sb;
    __shared__ float red[256];
    int t = threadIdx.x;
    if (t < 128) sW[t] = Wc[t];
    if (t == 0)  sb = bc[0];
    __syncthreads();
    int e = blockIdx.x * 256 + t;
    float s = 0.f;
    if (e < E) {
        int i = ei[e], j = ej[e];
        const float* gi = g + (long)i * cols;
        const float* gj = g + (long)j * cols;
        const float* hi = h + (long)i * cols;
        const float* hj = h + (long)j * cols;
        float z1 = sb, z2 = sb;
        for (int k = 0; k < cols; ++k) {
            z1 += gi[k] * sW[k] + hj[k] * sW[cols + k];
            z2 += hi[k] * sW[k] + gj[k] * sW[cols + k];
        }
        s = -(logsig(z1) + logsig(z2));
    }
    red[t] = s; __syncthreads();
    for (int st = 128; st > 0; st >>= 1) { if (t < st) red[t] += red[t + st]; __syncthreads(); }
    if (t == 0) atomicAdd(accum, red[0]);
}

// unsupervised contrast: block per row i; accum += log(diag / rowmean(exp(A_i . B_j / tau)))
__global__ __launch_bounds__(256) void k_unsup(
    const float* __restrict__ A, const float* __restrict__ B,
    float* accum, int n, int cols)
{
    __shared__ float arow[64];
    __shared__ float red[256];
    __shared__ float sdiag;
    int i = blockIdx.x, t = threadIdx.x;
    if (t < cols) arow[t] = A[(long)i * cols + t];
    __syncthreads();
    float s = 0.f;
    for (int j = t; j < n; j += 256) {
        const float* b = B + (long)j * cols;
        float d = 0.f;
        for (int k = 0; k < cols; ++k) d += arow[k] * b[k];
        float ev = expf(d / TAUf);
        s += ev;
        if (j == i) sdiag = ev;
    }
    red[t] = s; __syncthreads();
    for (int st = 128; st > 0; st >>= 1) { if (t < st) red[t] += red[t + st]; __syncthreads(); }
    if (t == 0) atomicAdd(accum, logf(sdiag / (red[0] / (float)n)));
}

__global__ __launch_bounds__(256) void k_rowsum(const float* __restrict__ mtx, float* rs, int m) {
    __shared__ float red[256];
    int r = blockIdx.x, t = threadIdx.x;
    float s = 0.f;
    for (int j = t; j < m; j += 256) s += mtx[(long)r * m + j];
    red[t] = s; __syncthreads();
    for (int st = 128; st > 0; st >>= 1) { if (t < st) red[t] += red[t + st]; __syncthreads(); }
    if (t == 0) rs[r] = red[0];
}

// supervised contrast: block per train row i
__global__ __launch_bounds__(256) void k_sup(
    const float* __restrict__ Ab, const float* __restrict__ Bb,
    const int* __restrict__ tidx, const float* __restrict__ intra,
    const float* __restrict__ inter, const float* __restrict__ rs,
    float* accum, int m, int cols)
{
    __shared__ float arow[64];
    __shared__ float redp[256];
    __shared__ float redn[256];
    int i = blockIdx.x, t = threadIdx.x;
    int gi = tidx[i];
    if (t < cols) arow[t] = Ab[(long)gi * cols + t];
    __syncthreads();
    float sp = 0.f, sn = 0.f;
    for (int j = t; j < m; j += 256) {
        const float* b = Bb + (long)tidx[j] * cols;
        float d = 0.f;
        for (int k = 0; k < cols; ++k) d += arow[k] * b[k];
        float hc = expf(d / TAUf);
        sp += hc * intra[(long)i * m + j];
        sn += hc * inter[(long)i * m + j];
    }
    redp[t] = sp; redn[t] = sn; __syncthreads();
    for (int st = 128; st > 0; st >>= 1) {
        if (t < st) { redp[t] += redp[t + st]; redn[t] += redn[t + st]; }
        __syncthreads();
    }
    if (t == 0) {
        float SP = redp[0];
        float SN = (redn[0] + SP) / (float)(m - 1);
        atomicAdd(accum, logf((SP / rs[i]) / SN));
    }
}

__global__ void k_sumsq(const float* __restrict__ p, long n, float* accum) {
    __shared__ float red[256];
    int t = threadIdx.x; float s = 0.f;
    for (long i = (long)blockIdx.x * 256 + t; i < n; i += (long)gridDim.x * 256) s += p[i] * p[i];
    red[t] = s; __syncthreads();
    for (int st = 128; st > 0; st >>= 1) { if (t < st) red[t] += red[t + st]; __syncthreads(); }
    if (t == 0) atomicAdd(accum, red[0]);
}

// sc: [0]=masksum [1]=ce_sum [2]=acc_sum [3]=edge_sum [4]=unsup_logsum
//     [5]=sup_logsum [6]=wd_sumsq
__global__ void k_finalize(const float* __restrict__ sc, float* __restrict__ outsc) {
    if (threadIdx.x == 0 && blockIdx.x == 0) {
        float loss = sc[1] / (float)Nn
                   + 0.4f * (sc[3] / (float)Ee)
                   - HP1f * (sc[4] / (float)(2 * Nn))
                   - HP1f * (sc[5] / (float)(2 * Mm))
                   + WDf * 0.5f * sc[6];
        outsc[0] = loss;
        outsc[1] = sc[2] / (float)Nn;
    }
}

// ---------------------------------------------------------------------------
// Host orchestration
// ---------------------------------------------------------------------------
extern "C" void kernel_launch(void* const* d_in, const int* in_sizes, int n_in,
                              void* d_out, int out_size, void* d_ws, size_t ws_size,
                              hipStream_t stream) {
    const float* features = (const float*)d_in[0];   // N x D
    const float* support  = (const float*)d_in[1];   // N x N
    const float* labels   = (const float*)d_in[2];   // N x C
    const float* intra    = (const float*)d_in[3];   // M x M
    const float* inter    = (const float*)d_in[4];   // M x M
    const float* W0  = (const float*)d_in[5];
    const float* b0  = (const float*)d_in[6];
    const float* W1  = (const float*)d_in[7];
    const float* b1  = (const float*)d_in[8];
    const float* Wh0 = (const float*)d_in[9];
    const float* bh0 = (const float*)d_in[10];
    const float* Wh1 = (const float*)d_in[11];
    const float* bh1 = (const float*)d_in[12];
    const float* Wc  = (const float*)d_in[13];
    const float* bc  = (const float*)d_in[14];
    const int* epi   = (const int*)d_in[15];
    const int* epj   = (const int*)d_in[16];
    const int* tidx  = (const int*)d_in[17];
    const int* mask  = (const int*)d_in[18];

    float* out = (float*)d_out;

    // workspace bump allocator (256B aligned)
    char* w = (char*)d_ws;
    auto alloc = [&](size_t bytes) -> char* {
        char* p = w;
        w += (bytes + 255) & ~(size_t)255;
        return p;
    };
    _Float16* sup16 = (_Float16*)alloc((size_t)Nn * Nn * 2);
    _Float16* x16   = (_Float16*)alloc((size_t)Nn * Dd * 2);
    _Float16* w0h   = (_Float16*)alloc((size_t)Dd * Hh * 2);
    _Float16* wh0h  = (_Float16*)alloc((size_t)Dd * Hh * 2);
    _Float16* w1h   = (_Float16*)alloc((size_t)Hh * Cc * 2);
    _Float16* wh1h  = (_Float16*)alloc((size_t)Hh * Cc * 2);
    float*    t0    = (float*)   alloc((size_t)Nn * Hh * 4);
    _Float16* t0h   = (_Float16*)alloc((size_t)Nn * Hh * 2);
    float*    h0    = (float*)   alloc((size_t)Nn * Hh * 4);
    _Float16* h0h   = (_Float16*)alloc((size_t)Nn * Hh * 2);
    float*    t1    = (float*)   alloc((size_t)Nn * Cc * 4);
    _Float16* t1h   = (_Float16*)alloc((size_t)Nn * Cc * 2);
    float*    h1    = (float*)   alloc((size_t)Nn * Cc * 4);
    float*    gcn   = (float*)   alloc((size_t)Nn * Cc * 4);
    float*    hgcn  = (float*)   alloc((size_t)Nn * Cc * 4);
    float*    rs    = (float*)   alloc((size_t)Mm * 4);
    float*    sc    = (float*)   alloc(16 * 4);
    float*    zbias = (float*)   alloc(256 * 4);

    dim3 blk(256);
    auto cvt = [&](const float* s, _Float16* d, long n) {
        k_cvt_f16<<<dim3((unsigned)((n + 255) / 256)), blk, 0, stream>>>(s, d, n);
    };

    // zero-bias vector for GEMMs without bias; scalar accumulators
    k_zero<<<dim3(1), blk, 0, stream>>>(zbias, 256);
    k_zero<<<dim3(1), dim3(16), 0, stream>>>(sc, 16);

    // --- f16 conversions (support fits L2 at 128MB in f16)
    cvt(support,  sup16, (long)Nn * Nn);
    cvt(features, x16,   (long)Nn * Dd);
    cvt(W0,  w0h,  (long)Dd * Hh);
    cvt(Wh0, wh0h, (long)Dd * Hh);
    cvt(W1,  w1h,  (long)Hh * Cc);
    cvt(Wh1, wh1h, (long)Hh * Cc);

    // --- GCN branch
    k_gemm_f16<0><<<dim3(Hh / BN, Nn / BM), blk, 0, stream>>>(x16, w0h, zbias, t0, Nn, Hh, Dd);
    cvt(t0, t0h, (long)Nn * Hh);
    k_gemm_f16<1><<<dim3(Hh / BN, Nn / BM), blk, 0, stream>>>(sup16, t0h, b0, h0, Nn, Hh, Nn);
    cvt(h0, h0h, (long)Nn * Hh);
    k_gemm_f16<0><<<dim3(Cc / BN, Nn / BM), blk, 0, stream>>>(h0h, w1h, zbias, t1, Nn, Cc, Hh);
    cvt(t1, t1h, (long)Nn * Cc);
    k_gemm_f16<0><<<dim3(Cc / BN, Nn / BM), blk, 0, stream>>>(sup16, t1h, b1, h1, Nn, Cc, Nn);
    k_l2norm<<<dim3(Nn), dim3(64), 0, stream>>>(h1, gcn, Cc);

    // --- HGCN branch (reuse intermediates)
    k_gemm_f16<0><<<dim3(Hh / BN, Nn / BM), blk, 0, stream>>>(x16, wh0h, zbias, t0, Nn, Hh, Dd);
    cvt(t0, t0h, (long)Nn * Hh);
    k_gemm_f16<1><<<dim3(Hh / BN, Nn / BM), blk, 0, stream>>>(sup16, t0h, bh0, h0, Nn, Hh, Nn);
    cvt(h0, h0h, (long)Nn * Hh);
    k_gemm_f16<0><<<dim3(Cc / BN, Nn / BM), blk, 0, stream>>>(h0h, wh1h, zbias, t1, Nn, Cc, Hh);
    cvt(t1, t1h, (long)Nn * Cc);
    k_gemm_f16<0><<<dim3(Cc / BN, Nn / BM), blk, 0, stream>>>(sup16, t1h, bh1, h1, Nn, Cc, Nn);
    k_l2norm<<<dim3(Nn), dim3(64), 0, stream>>>(h1, hgcn, Cc);

    // --- outputs = l2norm(0.6*gcn_n + 0.4*hgcn_n)
    k_combine<<<dim3(Nn), dim3(64), 0, stream>>>(gcn, hgcn, out, Cc);

    // --- scalar losses
    k_masksum<<<dim3(32), blk, 0, stream>>>(mask, sc, Nn);
    k_ce<<<dim3(Nn), dim3(64), 0, stream>>>(out, labels, mask, sc, &sc[1], &sc[2], Cc, Nn);
    k_edge<<<dim3(Ee / 256), blk, 0, stream>>>(gcn, hgcn, epi, epj, Wc, bc, &sc[3], Ee, Cc);
    k_unsup<<<dim3(Nn), blk, 0, stream>>>(gcn, hgcn, &sc[4], Nn, Cc);
    k_unsup<<<dim3(Nn), blk, 0, stream>>>(hgcn, gcn, &sc[4], Nn, Cc);
    k_rowsum<<<dim3(Mm), blk, 0, stream>>>(intra, rs, Mm);
    k_sup<<<dim3(Mm), blk, 0, stream>>>(gcn, hgcn, tidx, intra, inter, rs, &sc[5], Mm, Cc);
    k_sup<<<dim3(Mm), blk, 0, stream>>>(hgcn, gcn, tidx, intra, inter, rs, &sc[5], Mm, Cc);

    // weight decay (W0,b0,W1,b1,Wc,bc only, per reference)
    k_sumsq<<<dim3(32), blk, 0, stream>>>(W0, (long)Dd * Hh, &sc[6]);
    k_sumsq<<<dim3(1),  blk, 0, stream>>>(b0, (long)Hh,      &sc[6]);
    k_sumsq<<<dim3(8),  blk, 0, stream>>>(W1, (long)Hh * Cc, &sc[6]);
    k_sumsq<<<dim3(1),  blk, 0, stream>>>(b1, (long)Cc,      &sc[6]);
    k_sumsq<<<dim3(1),  blk, 0, stream>>>(Wc, (long)2 * Cc,  &sc[6]);
    k_sumsq<<<dim3(1),  blk, 0, stream>>>(bc, 1L,            &sc[6]);

    k_finalize<<<dim3(1), dim3(1), 0, stream>>>(sc, out + (size_t)Nn * Cc);
}
